// MessagePassing_9096740732969
// MI455X (gfx1250) — matrix operand order repaired and checked by reading, not compiled, not run on hardware
//
#include <hip/hip_runtime.h>
#include <stdint.h>

// MPNN message passing on gfx1250 (MI455X), wave32 + WMMA bf16.
//
//   msg = G @ EK'   where G[e][(k,j)] = bond[e][k] * h[dst[e]][j]   (per 16-edge tile:
//   17 K-chunks x 2 N-tiles = 34 x v_wmma_f32_16x16x32_bf16, 100% MAC utilization)
//   segment-sum via global_atomic_add_f32 (agg fits in 192MB L2)
//   GRU: two 16x32 @ 32x96 WMMA GEMMs per atom tile + VALU gates.
//   B-fragments are double-buffered from LDS so the per-chunk s_wait_dscnt does
//   not fully drain (loads for chunk c+1 in flight across chunk c's WMMAs).

#define N_ATOMS    50000
#define N_EDGES    300000
#define UNITS      32
#define BOND_DIM   16
#define STEPS      4
#define NCHUNK     17                 // 16 bond-dim chunks + 1 edge_bias chunk
#define EK_FRAGS   (NCHUNK * 2)       // 34 B-fragments (2 N-tiles of the 32 output units)
#define GK_FRAGS   12                 // 6 N-tiles x {gru_kernel, gru_rkernel}
#define EDGE_TILES (N_EDGES / 16)     // 18750 (exact)
#define ATOM_TILES (N_ATOMS / 16)     // 3125  (exact)

typedef __attribute__((ext_vector_type(16))) __bf16   v16bf;
typedef __attribute__((ext_vector_type(8)))  float    v8f;
typedef __attribute__((ext_vector_type(8)))  uint32_t v8u;

static __device__ __forceinline__ uint32_t pack2_bf16(float lo, float hi) {
  // bf16 pair, even element in [15:0] (truncation)
  uint32_t a = __builtin_bit_cast(uint32_t, lo);
  uint32_t b = __builtin_bit_cast(uint32_t, hi);
  return (a >> 16) | (b & 0xFFFF0000u);
}

static __device__ __forceinline__ uint32_t pk_mul_bf16(uint32_t a, uint32_t b) {
  uint32_t d;
  asm("v_pk_mul_bf16 %0, %1, %2" : "=v"(d) : "v"(a), "v"(b));
  return d;
}

static __device__ __forceinline__ v16bf frag_cast(v8u u) {
  return __builtin_bit_cast(v16bf, u);
}

static __device__ __forceinline__ v16bf frag_from_lds(const uint4* p) {
  uint4 t0 = p[0], t1 = p[1];
  v8u u;
  u[0] = t0.x; u[1] = t0.y; u[2] = t0.z; u[3] = t0.w;
  u[4] = t1.x; u[5] = t1.y; u[6] = t1.z; u[7] = t1.w;
  return __builtin_bit_cast(v16bf, u);
}

static __device__ __forceinline__ float fast_sigmoid(float x) {
  return 1.0f / (1.0f + __expf(-x));
}

// ---------------------------------------------------------------------------
// Prep: pre-swizzle EK' (512x32 + bias row-block) into per-lane B-fragments.
// Fragment f = chunk*2 + ntile. Within a fragment: lane L holds column
// i = ntile*16 + (L&15), K_local = (L>>4)*16 + q for q = 0..15 (pairs even-low).
// EK'[(k,j)][i] = edge_kernel[k][i*32 + j]; chunk 16 is edge_bias.
// ---------------------------------------------------------------------------
__global__ void prep_ek(const float* __restrict__ ek, const float* __restrict__ ebias,
                        uint16_t* __restrict__ out) {
  int idx = blockIdx.x * 256 + threadIdx.x;
  if (idx >= EK_FRAGS * 512) return;
  int f    = idx >> 9;
  int r    = idx & 511;
  int lane = r >> 4;
  int q    = r & 15;
  int c    = f >> 1;
  int t    = f & 1;
  int kl   = ((lane >> 4) << 4) + q;        // j within chunk (= K_local)
  int i    = (t << 4) + (lane & 15);        // output unit
  float val = (c < 16) ? ek[c * (UNITS * UNITS) + i * UNITS + kl]
                       : ebias[i * UNITS + kl];
  out[idx] = (uint16_t)(__builtin_bit_cast(uint32_t, val) >> 16);
}

// Same swizzle for the two GRU weight matrices (32 x 96 each), 6 N-tiles apiece.
__global__ void prep_gk(const float* __restrict__ gk, const float* __restrict__ rk,
                        uint16_t* __restrict__ out) {
  int idx = blockIdx.x * 256 + threadIdx.x;
  if (idx >= GK_FRAGS * 512) return;
  int f    = idx >> 9;
  int r    = idx & 511;
  int lane = r >> 4;
  int q    = r & 15;
  const float* src = (f < 6) ? gk : rk;
  int t    = (f < 6) ? f : f - 6;
  int kl   = ((lane >> 4) << 4) + q;        // input unit (K)
  int i    = (t << 4) + (lane & 15);        // output column in [0,96)
  out[idx] = (uint16_t)(__builtin_bit_cast(uint32_t, src[kl * 96 + i]) >> 16);
}

__global__ void zero_f(float* __restrict__ p, int n) {
  int i = blockIdx.x * blockDim.x + threadIdx.x;
  if (i < n) p[i] = 0.0f;
}

// ---------------------------------------------------------------------------
// Build a 16x32 bf16 A-fragment from a contiguous 32-float row.
// Lane layout (contiguous, matches prep swizzle): lane<16 -> K in {0..15}; lane>=16 -> {16..31}.
// ---------------------------------------------------------------------------
static __device__ __forceinline__ void load_rowA(const float* __restrict__ row,
                                                 int half, uint32_t nbp[8]) {
  const float4* hp = (const float4*)row;
  float4 f0 = hp[4 * half + 0];
  float4 f1 = hp[4 * half + 1];
  float4 f2 = hp[4 * half + 2];
  float4 f3 = hp[4 * half + 3];
  nbp[0] = pack2_bf16(f0.x, f0.y);  nbp[1] = pack2_bf16(f0.z, f0.w);
  nbp[2] = pack2_bf16(f1.x, f1.y);  nbp[3] = pack2_bf16(f1.z, f1.w);
  nbp[4] = pack2_bf16(f2.x, f2.y);  nbp[5] = pack2_bf16(f2.z, f2.w);
  nbp[6] = pack2_bf16(f3.x, f3.y);  nbp[7] = pack2_bf16(f3.z, f3.w);
}

// ---------------------------------------------------------------------------
// Edge-network + scatter: one 16-edge tile per wave iteration.
// ---------------------------------------------------------------------------
__global__ __launch_bounds__(1024) void edge_msg(
    const float* __restrict__ h, const float* __restrict__ bond,
    const int* __restrict__ pair, const uint16_t* __restrict__ ekf,
    float* __restrict__ agg) {
  __shared__ uint4 sEK[EK_FRAGS * 64];  // 34 KB, constant across all tiles/steps
  for (int i = threadIdx.x; i < EK_FRAGS * 64; i += blockDim.x)
    sEK[i] = ((const uint4*)ekf)[i];
  __syncthreads();

  const int wave   = (blockIdx.x * blockDim.x + threadIdx.x) >> 5;
  const int lane   = threadIdx.x & 31;
  const int half   = lane >> 4;
  const int el     = lane & 15;
  const int nwaves = (gridDim.x * blockDim.x) >> 5;
  const uint4* myEK = &sEK[lane * 2];   // lane-local base; fragment f at +f*64

  for (int tile = wave; tile < EDGE_TILES; tile += nwaves) {
    const int e    = tile * 16 + el;
    const int dsti = pair[2 * e + 1];

    // Prefetch next tile's index block while we compute this one.
    int nt = tile + nwaves;
    if (nt < EDGE_TILES) __builtin_prefetch(&pair[2 * nt * 16], 0, 1);

    // Neighbor features -> cached bf16 A-pairs (scaled per chunk below).
    uint32_t nbp[8];
    load_rowA(h + (size_t)dsti * UNITS, half, nbp);

    // All 16 bond features of this edge.
    const float4* bp = (const float4*)(bond + (size_t)e * BOND_DIM);
    float4 b0 = bp[0], b1 = bp[1], b2 = bp[2], b3 = bp[3];
    const float bf[16] = {b0.x, b0.y, b0.z, b0.w, b1.x, b1.y, b1.z, b1.w,
                          b2.x, b2.y, b2.z, b2.w, b3.x, b3.y, b3.z, b3.w};

    v8f acc0 = {0.f, 0.f, 0.f, 0.f, 0.f, 0.f, 0.f, 0.f};
    v8f acc1 = {0.f, 0.f, 0.f, 0.f, 0.f, 0.f, 0.f, 0.f};

    // Software pipeline: B fragments for chunk c+1 are loaded before the
    // WMMAs of chunk c, keeping 2x ds_load_b128 pairs in flight across each
    // pair of WMMAs (partial s_wait_dscnt instead of a full drain).
    v16bf Bc0 = frag_from_lds(myEK + 0 * 64);
    v16bf Bc1 = frag_from_lds(myEK + 1 * 64);

#pragma unroll
    for (int c = 0; c < NCHUNK; ++c) {
      v16bf Bn0, Bn1;
      if (c + 1 < NCHUNK) {
        Bn0 = frag_from_lds(myEK + ((c + 1) * 2 + 0) * 64);
        Bn1 = frag_from_lds(myEK + ((c + 1) * 2 + 1) * 64);
      }
      float s = (c < 16) ? bf[c] : 1.0f;           // chunk 16 == edge_bias term
      uint32_t sb = pack2_bf16(s, s);
      v8u av;
#pragma unroll
      for (int v = 0; v < 8; ++v) av[v] = pk_mul_bf16(nbp[v], sb);
      v16bf A = frag_cast(av);
      acc0 = __builtin_amdgcn_wmma_f32_16x16x32_bf16(false, A, false, Bc0,
                                                     (short)0, acc0, false, false);
      acc1 = __builtin_amdgcn_wmma_f32_16x16x32_bf16(false, A, false, Bc1,
                                                     (short)0, acc1, false, false);
      Bc0 = Bn0;
      Bc1 = Bn1;
    }

    // D layout: lane -> column (el / 16+el), VGPR r -> edge row (r + 8*half).
#pragma unroll
    for (int r = 0; r < 8; ++r) {
      int erow = tile * 16 + r + 8 * half;
      int srce = pair[2 * erow + 0];
      atomicAdd(&agg[(size_t)srce * UNITS + el],      acc0[r]);
      atomicAdd(&agg[(size_t)srce * UNITS + 16 + el], acc1[r]);
    }
  }
}

// ---------------------------------------------------------------------------
// GRU cell: one 16-atom tile per wave iteration.
// ---------------------------------------------------------------------------
__global__ __launch_bounds__(1024) void gru_step(
    const float* __restrict__ agg, const float* __restrict__ hin,
    const uint16_t* __restrict__ gkf, const float* __restrict__ gbias,
    float* __restrict__ hout) {
  __shared__ uint4 sGK[GK_FRAGS * 64];  // 12 KB
  for (int i = threadIdx.x; i < GK_FRAGS * 64; i += blockDim.x)
    sGK[i] = ((const uint4*)gkf)[i];
  __syncthreads();

  const int wave   = (blockIdx.x * blockDim.x + threadIdx.x) >> 5;
  const int lane   = threadIdx.x & 31;
  const int half   = lane >> 4;
  const int el     = lane & 15;
  const int nwaves = (gridDim.x * blockDim.x) >> 5;
  const uint4* myGK = &sGK[lane * 2];

  for (int tile = wave; tile < ATOM_TILES; tile += nwaves) {
    const int arow = tile * 16 + el;

    uint32_t xa[8], ha[8];
    load_rowA(agg + (size_t)arow * UNITS, half, xa);
    load_rowA(hin + (size_t)arow * UNITS, half, ha);
    v16bf Ax = frag_cast(*(v8u*)xa);
    v16bf Ah = frag_cast(*(v8u*)ha);

    // Double-buffered B fragments, same rationale as edge_msg.
    v16bf Bk = frag_from_lds(myGK + 0 * 64);
    v16bf Br = frag_from_lds(myGK + 6 * 64);

    v8f mx[6], mh[6];
#pragma unroll
    for (int t = 0; t < 6; ++t) {
      v16bf Bkn, Brn;
      if (t + 1 < 6) {
        Bkn = frag_from_lds(myGK + (t + 1) * 64);
        Brn = frag_from_lds(myGK + (7 + t) * 64);
      }
      v8f z = {0.f, 0.f, 0.f, 0.f, 0.f, 0.f, 0.f, 0.f};
      mx[t] = __builtin_amdgcn_wmma_f32_16x16x32_bf16(false, Ax, false, Bk,
                                                      (short)0, z, false, false);
      mh[t] = __builtin_amdgcn_wmma_f32_16x16x32_bf16(false, Ah, false, Br,
                                                      (short)0, z, false, false);
      Bk = Bkn;
      Br = Brn;
    }

    // Add biases (column c = 16t + el of the 96-wide output).
#pragma unroll
    for (int t = 0; t < 6; ++t) {
      float bx = gbias[t * 16 + el];
      float bh = gbias[96 + t * 16 + el];
#pragma unroll
      for (int r = 0; r < 8; ++r) { mx[t][r] += bx; mh[t][r] += bh; }
    }

    // Gates. Tiles {0,1}=z, {2,3}=r, {4,5}=h; unit u = p*16 + el.
#pragma unroll
    for (int r = 0; r < 8; ++r) {
      int atom = tile * 16 + r + 8 * half;
#pragma unroll
      for (int p = 0; p < 2; ++p) {
        int u = p * 16 + el;
        float z  = fast_sigmoid(mx[p][r]     + mh[p][r]);
        float rg = fast_sigmoid(mx[2 + p][r] + mh[2 + p][r]);
        float hh = tanhf(mx[4 + p][r] + rg * mh[4 + p][r]);
        float ho = hin[(size_t)atom * UNITS + u];
        hout[(size_t)atom * UNITS + u] = z * ho + (1.0f - z) * hh;
      }
    }
  }
}

// ---------------------------------------------------------------------------
extern "C" void kernel_launch(void* const* d_in, const int* in_sizes, int n_in,
                              void* d_out, int out_size, void* d_ws, size_t ws_size,
                              hipStream_t stream) {
  (void)in_sizes; (void)n_in; (void)out_size; (void)ws_size;
  const float* atomf = (const float*)d_in[0];
  const float* bond  = (const float*)d_in[1];
  const int*   pair  = (const int*)d_in[2];
  const float* ek    = (const float*)d_in[3];
  const float* ebias = (const float*)d_in[4];
  const float* gk    = (const float*)d_in[5];
  const float* rk    = (const float*)d_in[6];
  const float* gbias = (const float*)d_in[7];

  char* ws = (char*)d_ws;
  const size_t hbytes = (size_t)N_ATOMS * UNITS * sizeof(float);  // 6.4 MB
  float*    agg = (float*)ws;
  float*    hA  = (float*)(ws + hbytes);
  uint16_t* ekf = (uint16_t*)(ws + 2 * hbytes);
  uint16_t* gkf = (uint16_t*)(ws + 2 * hbytes + (size_t)EK_FRAGS * 512 * 2);
  float*    hB  = (float*)d_out;

  prep_ek<<<(EK_FRAGS * 512 + 255) / 256, 256, 0, stream>>>(ek, ebias, ekf);
  prep_gk<<<(GK_FRAGS * 512 + 255) / 256, 256, 0, stream>>>(gk, rk, gkf);

  const float* hcur = atomf;
  for (int s = 0; s < STEPS; ++s) {
    zero_f<<<(N_ATOMS * UNITS + 1023) / 1024, 1024, 0, stream>>>(agg, N_ATOMS * UNITS);
    edge_msg<<<128, 1024, 0, stream>>>(hcur, bond, pair, ekf, agg);
    float* hnext = (s & 1) ? hB : hA;   // step 3 (last) lands in d_out
    gru_step<<<98, 1024, 0, stream>>>(agg, hcur, gkf, gbias, hnext);
    hcur = hnext;
  }
}
